// AdaptiveCrossFrequencyModule_67980742361496
// MI455X (gfx1250) — matrix operand order repaired and checked by reading, not compiled, not run on hardware
//
#include <hip/hip_runtime.h>
#include <hip/hip_bf16.h>

typedef __bf16 bf16_t;
typedef __attribute__((ext_vector_type(16))) __bf16 v16bf;
typedef __attribute__((ext_vector_type(8)))  __bf16 v8bf;
typedef __attribute__((ext_vector_type(8)))  float  v8f;

#define WPB 8  // waves per block (256 threads, wave32)

// Build an A-fragment (contiguous K order, matching B):
// lane m=lane%16 holds K [(lane/16)*16 .. +16) -> one 32B load.
__device__ __forceinline__ v16bf load_afrag(const bf16_t* p)
{
    return *(const v16bf*)p;
}

// ---------------------------------------------------------------------------
// Init: 32x32 DCT-II matrix and radial low-frequency mask
// ---------------------------------------------------------------------------
__global__ __launch_bounds__(1024) void k_init(float* __restrict__ D,
                                               float* __restrict__ LM)
{
    int t = threadIdx.x;          // 0..1023
    int k = t >> 5, n = t & 31;
    const float PI = 3.14159265358979323846f;
    float v = __cosf(PI * (2.0f * n + 1.0f) * k / 64.0f) * sqrtf(2.0f / 32.0f);
    if (k == 0) v *= 0.70710678118654752440f;
    D[t] = v;
    float ry = (float)k / 31.0f, rx = (float)n / 31.0f;
    float rr = sqrtf(ry * ry + rx * rx) / 1.41421356237309504880f;
    LM[t] = (rr <= 0.35f) ? 1.0f : 0.0f;
}

// ---------------------------------------------------------------------------
// Elementwise f32 -> bf16
// ---------------------------------------------------------------------------
__global__ __launch_bounds__(256) void k_f32_to_bf16(const float* __restrict__ in,
                                                     bf16_t* __restrict__ out, long n)
{
    long i = (long)blockIdx.x * 256 + threadIdx.x;
    long stride = (long)gridDim.x * 256;
    for (; i < n; i += stride) out[i] = (bf16_t)in[i];
}

// ---------------------------------------------------------------------------
// Transpose+convert: f32 [b][R][S] -> bf16 [b][S][R]  (K-contiguous staging
// so every WMMA operand load is a contiguous 16B/32B per-lane access)
// ---------------------------------------------------------------------------
__global__ __launch_bounds__(256) void k_transpose_bf16(const float* __restrict__ in,
                                                        bf16_t* __restrict__ out,
                                                        int R, int S)
{
    __shared__ float tile[32][33];
    int bs = blockIdx.x * 32;   // S offset
    int br = blockIdx.y * 32;   // R offset
    int b  = blockIdx.z;
    const float* ip = in  + (long)b * R * S;
    bf16_t*      op = out + (long)b * R * S;
    int tx = threadIdx.x & 31, ty = threadIdx.x >> 5;  // ty in 0..7
    #pragma unroll
    for (int r = ty; r < 32; r += 8)
        tile[r][tx] = ip[(long)(br + r) * S + bs + tx];
    __syncthreads();
    #pragma unroll
    for (int r = ty; r < 32; r += 8)
        op[(long)(bs + r) * R + (br + tx)] = (bf16_t)tile[tx][r];
}

// ---------------------------------------------------------------------------
// Pack 3x3 conv weights [O][C][3][3] f32 -> [9][O][C] bf16 (tap-major)
// ---------------------------------------------------------------------------
__global__ __launch_bounds__(256) void k_packw3(const float* __restrict__ w,
                                                bf16_t* __restrict__ wp)
{
    int i = blockIdx.x * 256 + threadIdx.x; // o*512 + c
    if (i >= 512 * 512) return;
    #pragma unroll
    for (int t = 0; t < 9; ++t)
        wp[(long)t * 512 * 512 + i] = (bf16_t)w[(long)i * 9 + t];
}

// ---------------------------------------------------------------------------
// Generic batched WMMA GEMM, 2x2 register-blocked:
//   Out[b][i][j] = alpha * sum_k A[b][i][k] * Bt[b][j][k]  (+ bias[i])
// A, Bt bf16 (both row-major in K), Out f32. One 32x32 tile per wave:
// 2 A-fragments x 2 B-fragments -> 4 independent v_wmma_f32_16x16x32_bf16
// per K-step (hides XDL latency, halves operand bytes per WMMA vs 16x16).
// M and Nc must be multiples of 32 (true for every call site here).
// ---------------------------------------------------------------------------
__global__ __launch_bounds__(256) void k_gemm_bf16(
    const bf16_t* __restrict__ A,  long a_bs, int lda,
    const bf16_t* __restrict__ Bt, long b_bs, int ldb,
    float* __restrict__ Out,       long o_bs, int ldo,
    const float* __restrict__ bias, float alpha,
    int M, int Nc, int K, int nbatch)
{
    int lane = threadIdx.x & 31;
    int wave = threadIdx.x >> 5;
    int tilesM = M >> 5, tilesN = Nc >> 5;
    long tilesPerBatch = (long)tilesM * tilesN;
    long gwave = (long)blockIdx.x * WPB + wave;
    if (gwave >= tilesPerBatch * nbatch) return;   // wave-uniform: EXEC stays full
    int b  = (int)(gwave / tilesPerBatch);
    int t  = (int)(gwave % tilesPerBatch);
    int ti = t / tilesN;
    int tj = t % tilesN;

    const bf16_t* Ap0 = A  + (long)b * a_bs + (long)(ti * 32 + (lane & 15)) * lda + ((lane >> 4) << 4);
    const bf16_t* Ap1 = Ap0 + (long)16 * lda;
    const bf16_t* Bp0 = Bt + (long)b * b_bs + (long)(tj * 32 + (lane & 15)) * ldb + ((lane >> 4) << 4);
    const bf16_t* Bp1 = Bp0 + (long)16 * ldb;

    v8f acc00 = {}, acc01 = {}, acc10 = {}, acc11 = {};
    for (int k0 = 0; k0 < K; k0 += 32) {
        if (k0 + 256 < K) {                       // pull next lines toward WGP
            __builtin_prefetch(Ap0 + k0 + 256, 0, 3);
            __builtin_prefetch(Bp0 + k0 + 256, 0, 3);
        }
        v16bf a0 = load_afrag(Ap0 + k0);
        v16bf a1 = load_afrag(Ap1 + k0);
        v16bf b0 = *(const v16bf*)(Bp0 + k0);
        v16bf b1 = *(const v16bf*)(Bp1 + k0);
        acc00 = __builtin_amdgcn_wmma_f32_16x16x32_bf16(false, a0, false, b0, (short)0, acc00, false, false);
        acc01 = __builtin_amdgcn_wmma_f32_16x16x32_bf16(false, a0, false, b1, (short)0, acc01, false, false);
        acc10 = __builtin_amdgcn_wmma_f32_16x16x32_bf16(false, a1, false, b0, (short)0, acc10, false, false);
        acc11 = __builtin_amdgcn_wmma_f32_16x16x32_bf16(false, a1, false, b1, (short)0, acc11, false, false);
    }

    int j0  = tj * 32 + (lane & 15);
    int j1  = j0 + 16;
    int ib0 = ti * 32 + ((lane >> 4) << 3);
    float* Ob = Out + (long)b * o_bs;
    #pragma unroll
    for (int r = 0; r < 8; ++r) {
        float bz0 = bias ? bias[ib0 + r]      : 0.0f;
        float bz1 = bias ? bias[ib0 + 16 + r] : 0.0f;
        Ob[(long)(ib0 + r)      * ldo + j0] = acc00[r] * alpha + bz0;
        Ob[(long)(ib0 + r)      * ldo + j1] = acc01[r] * alpha + bz0;
        Ob[(long)(ib0 + 16 + r) * ldo + j0] = acc10[r] * alpha + bz1;
        Ob[(long)(ib0 + 16 + r) * ldo + j1] = acc11[r] * alpha + bz1;
    }
}

// ---------------------------------------------------------------------------
// 3x3 SAME conv as 9 shifted 1x1 convs on WMMA, 2x2 register-blocked.
//   Wp: [9][512][512] bf16 (tap-major packed weights)
//   Xt: [B][1024][512] bf16 (pixel-major, channel-contiguous)
//   Out: [B][512][1024] f32
// Wave computes 32 out-channels x 32 pixels. Border handling: per-lane
// zeroed B fragment for each pixel-column group.
// ---------------------------------------------------------------------------
__global__ __launch_bounds__(256) void k_conv3_bf16(
    const bf16_t* __restrict__ Wp, const bf16_t* __restrict__ Xt,
    float* __restrict__ Out, const float* __restrict__ bias)
{
    const int C = 512, N = 1024;
    int lane = threadIdx.x & 31;
    int wave = threadIdx.x >> 5;
    int gwave = blockIdx.x * WPB + wave;
    const int tilesN = N / 32;                   // 32
    const int tilesPerBatch = (C / 32) * tilesN; // 512
    int b  = gwave / tilesPerBatch;
    int t  = gwave % tilesPerBatch;
    int ti = t / tilesN;
    int tj = t % tilesN;

    int j0 = tj * 32 + (lane & 15);              // output pixel, group 0
    int j1 = j0 + 16;                            // output pixel, group 1
    int y0 = j0 >> 5, x0 = j0 & 31;
    int y1 = j1 >> 5, x1 = j1 & 31;
    int kb = (lane >> 4) << 4;
    int ka = (lane >> 4) << 4;

    const bf16_t* Xb    = Xt + (long)b * N * C;
    const bf16_t* Arow0 = Wp + (long)(ti * 32 + (lane & 15)) * C + ka;

    v16bf zero;
    #pragma unroll
    for (int e = 0; e < 16; ++e) zero[e] = (bf16_t)0.0f;

    v8f acc00 = {}, acc01 = {}, acc10 = {}, acc11 = {};
    for (int tap = 0; tap < 9; ++tap) {
        int dy = tap / 3 - 1, dx = tap % 3 - 1;
        int yy0 = y0 + dy, xx0 = x0 + dx;
        int yy1 = y1 + dy, xx1 = x1 + dx;
        bool v0 = (yy0 >= 0) && (yy0 < 32) && (xx0 >= 0) && (xx0 < 32);
        bool v1 = (yy1 >= 0) && (yy1 < 32) && (xx1 >= 0) && (xx1 < 32);
        int jp0 = ((yy0 << 5) | (xx0 & 31)) & 1023;   // clamped, safe address
        int jp1 = ((yy1 << 5) | (xx1 & 31)) & 1023;
        const bf16_t* B0 = Xb + (long)jp0 * C + kb;
        const bf16_t* B1 = Xb + (long)jp1 * C + kb;
        const bf16_t* Ar = Arow0 + (long)tap * C * C;
        for (int k0 = 0; k0 < C; k0 += 32) {
            v16bf a0 = load_afrag(Ar + k0);
            v16bf a1 = load_afrag(Ar + (long)16 * C + k0);
            v16bf b0 = v0 ? *(const v16bf*)(B0 + k0) : zero;
            v16bf b1 = v1 ? *(const v16bf*)(B1 + k0) : zero;
            acc00 = __builtin_amdgcn_wmma_f32_16x16x32_bf16(false, a0, false, b0, (short)0, acc00, false, false);
            acc01 = __builtin_amdgcn_wmma_f32_16x16x32_bf16(false, a0, false, b1, (short)0, acc01, false, false);
            acc10 = __builtin_amdgcn_wmma_f32_16x16x32_bf16(false, a1, false, b0, (short)0, acc10, false, false);
            acc11 = __builtin_amdgcn_wmma_f32_16x16x32_bf16(false, a1, false, b1, (short)0, acc11, false, false);
        }
    }
    int ib0 = ti * 32 + ((lane >> 4) << 3);
    float* Ob = Out + (long)b * C * N;
    #pragma unroll
    for (int r = 0; r < 8; ++r) {
        float bz0 = bias[ib0 + r];
        float bz1 = bias[ib0 + 16 + r];
        Ob[(long)(ib0 + r)      * N + j0] = acc00[r] + bz0;
        Ob[(long)(ib0 + r)      * N + j1] = acc01[r] + bz0;
        Ob[(long)(ib0 + 16 + r) * N + j0] = acc10[r] + bz1;
        Ob[(long)(ib0 + 16 + r) * N + j1] = acc11[r] + bz1;
    }
}

// ---------------------------------------------------------------------------
// Mean over H*W=1024 per (b,c) row: one wave per row
// ---------------------------------------------------------------------------
__global__ __launch_bounds__(256) void k_mean(const float* __restrict__ in,
                                              float* __restrict__ pooled, int rows)
{
    int lane = threadIdx.x & 31, wave = threadIdx.x >> 5;
    int row = blockIdx.x * WPB + wave;
    if (row >= rows) return;
    const float* p = in + (long)row * 1024;
    float s = 0.0f;
    for (int i = lane; i < 1024; i += 32) s += p[i];
    for (int off = 16; off > 0; off >>= 1) s += __shfl_down(s, off, 32);
    if (lane == 0) pooled[row] = s * (1.0f / 1024.0f);
}

// ---------------------------------------------------------------------------
// Small 2-layer MLP: out = sigmoid(W2 * relu(W1*pooled + b1) + b2)
// One block per batch. mid <= 128.
// ---------------------------------------------------------------------------
__global__ __launch_bounds__(512) void k_mlp(
    const float* __restrict__ w1, const float* __restrict__ b1,
    const float* __restrict__ w2, const float* __restrict__ b2,
    const float* __restrict__ pooled, float* __restrict__ out,
    int in_dim, int mid, int out_dim)
{
    __shared__ float h[128];
    int b = blockIdx.x, t = threadIdx.x;
    const float* pb = pooled + (long)b * in_dim;
    if (t < mid) {
        float s = b1[t];
        for (int c = 0; c < in_dim; ++c) s += w1[t * in_dim + c] * pb[c];
        h[t] = fmaxf(s, 0.0f);
    }
    __syncthreads();
    if (t < out_dim) {
        float s = b2[t];
        for (int m = 0; m < mid; ++m) s += w2[t * mid + m] * h[m];
        out[(long)b * out_dim + t] = 1.0f / (1.0f + expf(-s));
    }
}

// ---------------------------------------------------------------------------
// Frequency transform per (b,c) 32x32 tile, fully in LDS:
//   Y = D*X*D^T ; Y *= (gl*lm + gh*(1-lm)) ; Out = D^T*Y*D
// ---------------------------------------------------------------------------
__global__ __launch_bounds__(256) void k_freq(
    const float* __restrict__ in, float* __restrict__ out,
    const float* __restrict__ gates, const float* __restrict__ Dg,
    const float* __restrict__ LMg)
{
    __shared__ float Ds[1024], LMs[1024], X[1024], T[1024];
    int bc = blockIdx.x;            // b*512 + c
    int b = bc >> 9;
    const float* src = in + (long)bc * 1024;
    float* dst = out + (long)bc * 1024;
    for (int e = threadIdx.x; e < 1024; e += 256) {
        Ds[e] = Dg[e]; LMs[e] = LMg[e]; X[e] = src[e];
    }
    float gl = gates[b * 2 + 0];
    float gh = gates[b * 2 + 1];
    __syncthreads();
    for (int e = threadIdx.x; e < 1024; e += 256) {       // T = D * X
        int i = e >> 5, w = e & 31;
        float s = 0.0f;
        for (int h = 0; h < 32; ++h) s += Ds[i * 32 + h] * X[h * 32 + w];
        T[e] = s;
    }
    __syncthreads();
    for (int e = threadIdx.x; e < 1024; e += 256) {       // X = (T*D^T) .* mask
        int i = e >> 5, jj = e & 31;
        float s = 0.0f;
        for (int w = 0; w < 32; ++w) s += T[i * 32 + w] * Ds[jj * 32 + w];
        float m = LMs[e];
        X[e] = s * (gl * m + gh * (1.0f - m));
    }
    __syncthreads();
    for (int e = threadIdx.x; e < 1024; e += 256) {       // T = D^T * X
        int hh = e >> 5, jj = e & 31;
        float s = 0.0f;
        for (int i = 0; i < 32; ++i) s += Ds[i * 32 + hh] * X[i * 32 + jj];
        T[e] = s;
    }
    __syncthreads();
    for (int e = threadIdx.x; e < 1024; e += 256) {       // out = T * D
        int hh = e >> 5, w = e & 31;
        float s = 0.0f;
        for (int jj = 0; jj < 32; ++jj) s += T[hh * 32 + jj] * Ds[jj * 32 + w];
        dst[e] = s;
    }
}

// ---------------------------------------------------------------------------
// Row softmax over 1024 elements (in-place), one block per row
// ---------------------------------------------------------------------------
__global__ __launch_bounds__(256) void k_softmax(float* __restrict__ S)
{
    __shared__ float red[256];
    float* row = S + (long)blockIdx.x * 1024;
    int t = threadIdx.x;
    float x0 = row[t], x1 = row[t + 256], x2 = row[t + 512], x3 = row[t + 768];
    float m = fmaxf(fmaxf(x0, x1), fmaxf(x2, x3));
    red[t] = m; __syncthreads();
    for (int o = 128; o > 0; o >>= 1) { if (t < o) red[t] = fmaxf(red[t], red[t + o]); __syncthreads(); }
    float mm = red[0]; __syncthreads();
    float e0 = expf(x0 - mm), e1 = expf(x1 - mm), e2 = expf(x2 - mm), e3 = expf(x3 - mm);
    red[t] = e0 + e1 + e2 + e3; __syncthreads();
    for (int o = 128; o > 0; o >>= 1) { if (t < o) red[t] += red[t + o]; __syncthreads(); }
    float inv = 1.0f / red[0];
    row[t] = e0 * inv; row[t + 256] = e1 * inv; row[t + 512] = e2 * inv; row[t + 768] = e3 * inv;
}

// ---------------------------------------------------------------------------
// Final: out = rgb_p + spatial * ca_weight[b][c]
// ---------------------------------------------------------------------------
__global__ __launch_bounds__(256) void k_final(
    const float* __restrict__ rgbp, const float* __restrict__ spat,
    const float* __restrict__ caw, float* __restrict__ out, long n)
{
    long i = (long)blockIdx.x * 256 + threadIdx.x;
    if (i < n) out[i] = rgbp[i] + spat[i] * caw[i >> 10];
}

// ---------------------------------------------------------------------------
extern "C" void kernel_launch(void* const* d_in, const int* in_sizes, int n_in,
                              void* d_out, int out_size, void* d_ws, size_t ws_size,
                              hipStream_t stream)
{
    (void)in_sizes; (void)n_in; (void)out_size; (void)ws_size;
    const int  B = 16, C = 512, N = 1024, MID = 128;
    const long CN  = (long)C * N;        // 524288
    const long NN  = (long)N * N;        // 1048576
    const long BCN = (long)B * CN;       // 8388608
    const long BNN = (long)B * NN;       // 16777216

    const float* rgb   = (const float*)d_in[0];
    const float* dsm   = (const float*)d_in[1];
    const float* rgb_w = (const float*)d_in[2];
    const float* rgb_b = (const float*)d_in[3];
    const float* dsm_w = (const float*)d_in[4];
    const float* dsm_b = (const float*)d_in[5];
    const float* gw1   = (const float*)d_in[6];
    const float* gb1   = (const float*)d_in[7];
    const float* gw2   = (const float*)d_in[8];
    const float* gb2   = (const float*)d_in[9];
    const float* q_w   = (const float*)d_in[10];
    const float* q_b   = (const float*)d_in[11];
    const float* kw_   = (const float*)d_in[12];
    const float* k_b   = (const float*)d_in[13];
    const float* v_w   = (const float*)d_in[14];
    const float* v_b   = (const float*)d_in[15];
    const float* sp_w  = (const float*)d_in[16];
    const float* sp_b  = (const float*)d_in[17];
    const float* cw1   = (const float*)d_in[18];
    const float* cb1   = (const float*)d_in[19];
    const float* cw2   = (const float*)d_in[20];
    const float* cb2   = (const float*)d_in[21];
    float* outp = (float*)d_out;

    char* base = (char*)d_ws;
    size_t off = 0;
    auto alloc = [&](size_t bytes) -> void* {
        void* p = base + off;
        off += (bytes + 255) & ~(size_t)255;
        return p;
    };

    float*  Dm      = (float*)alloc(1024 * 4);
    float*  LMm     = (float*)alloc(1024 * 4);
    float*  pooled  = (float*)alloc((size_t)B * C * 4);
    float*  gates_r = (float*)alloc((size_t)B * 2 * 4);
    float*  gates_d = (float*)alloc((size_t)B * 2 * 4);
    float*  caw     = (float*)alloc((size_t)B * C * 4);
    bf16_t* wbf_rgb = (bf16_t*)alloc((size_t)C * C * 2);
    bf16_t* wbf_dsm = (bf16_t*)alloc((size_t)C * C * 2);
    bf16_t* wbf_q   = (bf16_t*)alloc((size_t)C * C * 2);
    bf16_t* wbf_k   = (bf16_t*)alloc((size_t)C * C * 2);
    bf16_t* wbf_v   = (bf16_t*)alloc((size_t)C * C * 2);
    bf16_t* wpack   = (bf16_t*)alloc((size_t)9 * C * C * 2);
    float*  P_RGB   = (float*)alloc((size_t)BCN * 4);  // rgb_p (live to end)
    float*  P_DSM   = (float*)alloc((size_t)BCN * 4);  // dsm_p -> attn output
    bf16_t* T1      = (bf16_t*)alloc((size_t)BCN * 2); // rgbT -> rgbfT -> qT -> attoutT
    bf16_t* T2      = (bf16_t*)alloc((size_t)BCN * 2); // dsmT -> dsmfT -> kT
    float*  FR      = (float*)alloc((size_t)BCN * 4);  // rgb_f -> q
    float*  FD      = (float*)alloc((size_t)BCN * 4);  // dsm_f -> k
    float*  Vf      = (float*)alloc((size_t)BCN * 4);  // v -> spatial
    float*  Sc      = (float*)alloc((size_t)BNN * 4);  // scores/attn
    bf16_t* ABF     = (bf16_t*)alloc((size_t)BNN * 2); // attn bf16
    bf16_t* VBF     = (bf16_t*)alloc((size_t)BCN * 2); // v bf16

    const int gConv1 = (int)((B * (C / 32) * (N / 32) + WPB - 1) / WPB);  // 1024
    const int gScore = (int)((B * (N / 32) * (N / 32) + WPB - 1) / WPB);  // 2048
    dim3 tg(N / 32, C / 32, B);

    k_init<<<1, 1024, 0, stream>>>(Dm, LMm);

    // weight conversions
    k_f32_to_bf16<<<(C * C + 255) / 256, 256, 0, stream>>>(rgb_w, wbf_rgb, (long)C * C);
    k_f32_to_bf16<<<(C * C + 255) / 256, 256, 0, stream>>>(dsm_w, wbf_dsm, (long)C * C);
    k_f32_to_bf16<<<(C * C + 255) / 256, 256, 0, stream>>>(q_w,   wbf_q,   (long)C * C);
    k_f32_to_bf16<<<(C * C + 255) / 256, 256, 0, stream>>>(kw_,   wbf_k,   (long)C * C);
    k_f32_to_bf16<<<(C * C + 255) / 256, 256, 0, stream>>>(v_w,   wbf_v,   (long)C * C);
    k_packw3<<<(C * C + 255) / 256, 256, 0, stream>>>(sp_w, wpack);

    // stage inputs pixel-major bf16, then 1x1 input projections (WMMA)
    k_transpose_bf16<<<tg, 256, 0, stream>>>(rgb, T1, C, N);
    k_transpose_bf16<<<tg, 256, 0, stream>>>(dsm, T2, C, N);
    k_gemm_bf16<<<gConv1, 256, 0, stream>>>(wbf_rgb, 0L, C, T1, CN, C, P_RGB, CN, N, rgb_b, 1.0f, C, N, C, B);
    k_gemm_bf16<<<gConv1, 256, 0, stream>>>(wbf_dsm, 0L, C, T2, CN, C, P_DSM, CN, N, dsm_b, 1.0f, C, N, C, B);

    // gate MLPs (per-batch scalars)
    k_mean<<<(B * C + WPB - 1) / WPB, 256, 0, stream>>>(P_RGB, pooled, B * C);
    k_mlp<<<B, 512, 0, stream>>>(gw1, gb1, gw2, gb2, pooled, gates_r, C, MID, 2);
    k_mean<<<(B * C + WPB - 1) / WPB, 256, 0, stream>>>(P_DSM, pooled, B * C);
    k_mlp<<<B, 512, 0, stream>>>(gw1, gb1, gw2, gb2, pooled, gates_d, C, MID, 2);

    // DCT gating (f32, LDS)
    k_freq<<<B * C, 256, 0, stream>>>(P_RGB, FR, gates_r, Dm, LMm);
    k_freq<<<B * C, 256, 0, stream>>>(P_DSM, FD, gates_d, Dm, LMm);

    // q/k/v projections (WMMA)
    k_transpose_bf16<<<tg, 256, 0, stream>>>(FR, T1, C, N);
    k_transpose_bf16<<<tg, 256, 0, stream>>>(FD, T2, C, N);
    k_gemm_bf16<<<gConv1, 256, 0, stream>>>(wbf_q, 0L, C, T1, CN, C, FR, CN, N, q_b, 1.0f, C, N, C, B);
    k_gemm_bf16<<<gConv1, 256, 0, stream>>>(wbf_k, 0L, C, T2, CN, C, FD, CN, N, k_b, 1.0f, C, N, C, B);
    k_gemm_bf16<<<gConv1, 256, 0, stream>>>(wbf_v, 0L, C, T2, CN, C, Vf, CN, N, v_b, 1.0f, C, N, C, B);

    // attention: scores = q^T k / sqrt(C)  (WMMA, both operands [N][C] bf16)
    k_transpose_bf16<<<tg, 256, 0, stream>>>(FR, T1, C, N);   // qT
    k_transpose_bf16<<<tg, 256, 0, stream>>>(FD, T2, C, N);   // kT
    k_f32_to_bf16<<<32768, 256, 0, stream>>>(Vf, VBF, BCN);
    float alpha = 1.0f / sqrtf((float)C);
    k_gemm_bf16<<<gScore, 256, 0, stream>>>(T1, CN, C, T2, CN, C, Sc, NN, N, nullptr, alpha, N, N, C, B);

    k_softmax<<<B * N, 256, 0, stream>>>(Sc);
    k_f32_to_bf16<<<65536, 256, 0, stream>>>(Sc, ABF, BNN);

    // out = attn @ v  (WMMA: A=v [C][m], Bt=attn [n][m])
    k_gemm_bf16<<<gConv1, 256, 0, stream>>>(VBF, CN, N, ABF, NN, N, P_DSM, CN, N, nullptr, 1.0f, C, N, N, B);

    // 3x3 spatial conv (WMMA, implicit GEMM over 9 taps, 2x2 blocked)
    k_transpose_bf16<<<tg, 256, 0, stream>>>(P_DSM, T1, C, N);
    k_conv3_bf16<<<gConv1, 256, 0, stream>>>(wpack, T1, Vf, sp_b);

    // channel attention + residual
    k_mean<<<(B * C + WPB - 1) / WPB, 256, 0, stream>>>(Vf, pooled, B * C);
    k_mlp<<<B, 512, 0, stream>>>(cw1, cb1, cw2, cb2, pooled, caw, C, MID, C);
    k_final<<<32768, 256, 0, stream>>>(P_RGB, Vf, caw, outp, BCN);
}
